// EncoderHead_19430432047109
// MI455X (gfx1250) — matrix operand order repaired and checked by reading, not compiled, not run on hardware
//
#include <hip/hip_runtime.h>
#include <hip/hip_bf16.h>

#define Bc 8
#define Tc 2048
#define Cc 768
#define Hc 64

typedef __attribute__((ext_vector_type(16))) __bf16 v16bf;
typedef __attribute__((ext_vector_type(8)))  float  v8f;

#define WMMA_BF16(a, b, c) \
  __builtin_amdgcn_wmma_f32_16x16x32_bf16(false, (a), false, (b), (short)0, (c), false, false)

union V16U { v16bf v; uint4 q[2]; };

// A-operand (16-bit 16x32): 16 contiguous K values per lane from p0 (p1 unused).
__device__ __forceinline__ v16bf ldA(const __bf16* p0, const __bf16* p1) {
  V16U u;
  u.q[0] = *reinterpret_cast<const uint4*>(p0);
  u.q[1] = *reinterpret_cast<const uint4*>(p0 + 8);
  return u.v;
}
// B-operand (16-bit 32x16): 16 contiguous K values per lane.
__device__ __forceinline__ v16bf ldB(const __bf16* p) {
  V16U u;
  u.q[0] = *reinterpret_cast<const uint4*>(p);
  u.q[1] = *reinterpret_cast<const uint4*>(p + 8);
  return u.v;
}
// A-operand built from fp32 source (on-the-fly bf16 convert).
__device__ __forceinline__ v16bf cvtA(const float* p0, const float* p1) {
  v16bf r;
#pragma unroll
  for (int i = 0; i < 8; ++i) r[i] = (__bf16)p0[i];
#pragma unroll
  for (int i = 0; i < 8; ++i) r[i + 8] = (__bf16)p0[i + 8];
  return r;
}

// Async Global -> LDS 16B copy (per lane), tracked by ASYNCcnt.
__device__ __forceinline__ void async_b128(unsigned ldsb, unsigned long long gaddr) {
  asm volatile("global_load_async_to_lds_b128 %0, %1, off"
               :: "v"(ldsb), "v"(gaddr) : "memory");
}
__device__ __forceinline__ void wait_async0() {
  asm volatile("s_wait_asynccnt 0x0" ::: "memory");
}
__device__ __forceinline__ unsigned lds_lo32(const void* p) {
  return (unsigned)(unsigned long long)(uintptr_t)p;  // LDS aperture: low 32 bits = LDS offset
}

// ---------------------------------------------------------------------------
// Kernel 1: WT[m][h][c] = bf16(W_m[c][h]), m: 0=Wq, 1=Wk, 2=Wv
// ---------------------------------------------------------------------------
__global__ void wt_kernel(const float* __restrict__ Wq, const float* __restrict__ Wk,
                          const float* __restrict__ Wv, __bf16* __restrict__ WT) {
  int idx = blockIdx.x * blockDim.x + threadIdx.x;
  if (idx >= 3 * Hc * Cc) return;
  int m = idx / (Hc * Cc);
  int h = (idx / Cc) % Hc;
  int c = idx % Cc;
  const float* W = (m == 0) ? Wq : (m == 1) ? Wk : Wv;
  WT[idx] = (__bf16)W[c * Hc + h];
}

// ---------------------------------------------------------------------------
// Kernel 2: q/k/v projection. 1 block = one (b, 16-row) tile, wave w = h-tile w.
// Writes q,k row-major bf16 [B,T,H]; v transposed bf16 [B,H,T].
// ---------------------------------------------------------------------------
__global__ __launch_bounds__(128)
void proj_kernel(const float* __restrict__ x, const __bf16* __restrict__ WT,
                 __bf16* __restrict__ q, __bf16* __restrict__ k, __bf16* __restrict__ vT) {
  const int wave = threadIdx.x >> 5;
  const int lane = threadIdx.x & 31;
  const int half = lane >> 4;
  const int l16  = lane & 15;
  const int kl   = half * 16;   // A-operand K base for this lane
  const int kb16 = half * 16;   // B-operand K base for this lane

  const int b  = blockIdx.x >> 7;          // T/16 = 128 tiles per batch
  const int t0 = (blockIdx.x & 127) << 4;

  const float*  xrow = x + (long long)(b * Tc + t0 + l16) * Cc;
  const __bf16* WTq  = WT + (long long)(16 * wave + l16) * Cc;          // q cols
  const __bf16* WTk  = WTq + (long long)Hc * Cc;
  const __bf16* WTv  = WTk + (long long)Hc * Cc;

  v8f accq = {}, acck = {}, accv = {};
#pragma unroll 4
  for (int K0 = 0; K0 < Cc; K0 += 32) {
    v16bf a  = cvtA(xrow + K0 + kl, xrow + K0 + kl + 16);
    v16bf bq = ldB(WTq + K0 + kb16);
    v16bf bk = ldB(WTk + K0 + kb16);
    v16bf bv = ldB(WTv + K0 + kb16);
    accq = WMMA_BF16(a, bq, accq);
    acck = WMMA_BF16(a, bk, acck);
    accv = WMMA_BF16(a, bv, accv);
  }

#pragma unroll
  for (int r = 0; r < 8; ++r) {
    int M = r + half * 8;
    long long o = (long long)(b * Tc + t0 + M) * Hc + 16 * wave + l16;
    q[o] = (__bf16)accq[r];
    k[o] = (__bf16)acck[r];
    vT[((long long)b * Hc + 16 * wave + l16) * Tc + t0 + M] = (__bf16)accv[r];
  }
}

// ---------------------------------------------------------------------------
// Kernel 3: fused flash attention. 1 wave = one (b, 16-query) tile; the 4
// waves of a block share double-buffered K/V tiles staged via async->LDS.
// ---------------------------------------------------------------------------
__global__ __launch_bounds__(128)
void attn_kernel(const __bf16* __restrict__ q, const __bf16* __restrict__ k,
                 const __bf16* __restrict__ vT, const int* __restrict__ mask,
                 float* __restrict__ out) {
  __shared__ __attribute__((aligned(16))) __bf16 Klds[2][32][64];  // [key][h]
  __shared__ __attribute__((aligned(16))) __bf16 Vlds[2][64][32];  // [h][key]
  __shared__ __attribute__((aligned(16))) __bf16 Plds[4][16][32];

  const int tid  = threadIdx.x;
  const int wave = tid >> 5;
  const int lane = tid & 31;
  const int half = lane >> 4;
  const int l16  = lane & 15;
  const int kl   = half * 16;
  const int kb16 = half * 16;

  const int qt = blockIdx.x * 4 + wave;    // all 4 waves: same batch b
  const int b  = qt >> 7;
  const int t0 = (qt & 127) << 4;

  const __bf16* kbase = k  + (long long)b * Tc * Hc;
  const __bf16* vbase = vT + (long long)b * Hc * Tc;
  const int*    mrow  = mask + b * Tc;

  // Stage one 32-key K tile (4KB, contiguous) + V tile (64 rows x 64B) into LDS.
  auto stage = [&](int buf, int kb) {
    // K: tile is contiguous [kb*32*Hc, +4KB); thread copies 32B
    unsigned long long ks = (unsigned long long)(uintptr_t)(kbase + (long long)kb * 32 * Hc) + tid * 32;
    unsigned kd = lds_lo32(&Klds[buf][0][0]) + tid * 32;
    async_b128(kd, ks);
    async_b128(kd + 16, ks + 16);
    // V: row h = tid>>1, 32B segment seg = tid&1
    int h = tid >> 1, seg = tid & 1;
    unsigned long long vs =
        (unsigned long long)(uintptr_t)(vbase + (long long)h * Tc + kb * 32 + seg * 16);
    unsigned vd = lds_lo32(&Vlds[buf][h][seg * 16]);
    async_b128(vd, vs);
    async_b128(vd + 16, vs + 16);
  };

  // q A-operands (16x64 split into two K=32 halves)
  const __bf16* qrow = q + (long long)(b * Tc + t0 + l16) * Hc;
  v16bf qa0 = ldA(qrow + kl,      qrow + kl + 16);
  v16bf qa1 = ldA(qrow + 32 + kl, qrow + 32 + kl + 16);

  v8f acc[4] = {};
  float m[8], l[8];
#pragma unroll
  for (int r = 0; r < 8; ++r) { m[r] = -__builtin_inff(); l[r] = 0.f; }

  const float scale = 0.03608439182435161f;  // C^-0.5 = 1/sqrt(768), per source

  stage(0, 0);
  wait_async0();
  __syncthreads();

  for (int kb = 0; kb < Tc / 32; ++kb) {
    const int cur = kb & 1;
    if (kb + 1 < Tc / 32) stage(cur ^ 1, kb + 1);  // overlap next tile with compute

    float p[2][8];
    // ---- S = q . k^T for 32 keys (two 16-key subtiles) ----
#pragma unroll
    for (int j = 0; j < 2; ++j) {
      const __bf16* krow = &Klds[cur][j * 16 + l16][0];
      v16bf b0 = ldB(krow + kb16);
      v16bf b1 = ldB(krow + 32 + kb16);
      v8f s = {};
      s = WMMA_BF16(qa0, b0, s);
      s = WMMA_BF16(qa1, b1, s);
      const int mk = mrow[kb * 32 + j * 16 + l16];  // this lane's key column
#pragma unroll
      for (int r = 0; r < 8; ++r)
        p[j][r] = (mk == 0) ? -__builtin_inff() : s[r] * scale;
    }

    // ---- online softmax (row reductions across the 16-lane half) ----
    float rmax[8];
#pragma unroll
    for (int r = 0; r < 8; ++r) rmax[r] = fmaxf(p[0][r], p[1][r]);
#pragma unroll
    for (int off = 1; off < 16; off <<= 1)
#pragma unroll
      for (int r = 0; r < 8; ++r) rmax[r] = fmaxf(rmax[r], __shfl_xor(rmax[r], off, 32));

    float rsum[8];
#pragma unroll
    for (int r = 0; r < 8; ++r) {
      float mnew  = fmaxf(m[r], rmax[r]);
      float alpha = (m[r] == -__builtin_inff()) ? 0.f : __expf(m[r] - mnew);
      float p0 = (mnew == -__builtin_inff()) ? 0.f : __expf(p[0][r] - mnew);
      float p1 = (mnew == -__builtin_inff()) ? 0.f : __expf(p[1][r] - mnew);
      p[0][r] = p0; p[1][r] = p1;
      rsum[r] = p0 + p1;
      m[r] = mnew;
      l[r] *= alpha;
      acc[0][r] *= alpha; acc[1][r] *= alpha; acc[2][r] *= alpha; acc[3][r] *= alpha;
    }
#pragma unroll
    for (int off = 1; off < 16; off <<= 1)
#pragma unroll
      for (int r = 0; r < 8; ++r) rsum[r] += __shfl_xor(rsum[r], off, 32);
#pragma unroll
    for (int r = 0; r < 8; ++r) l[r] += rsum[r];

    // ---- re-swizzle P (C-layout -> A-layout) through LDS ----
#pragma unroll
    for (int j = 0; j < 2; ++j)
#pragma unroll
      for (int r = 0; r < 8; ++r)
        Plds[wave][r + half * 8][j * 16 + l16] = (__bf16)p[j][r];
    asm volatile("s_wait_dscnt 0x0" ::: "memory");
    const __bf16* Pr = &Plds[wave][l16][0];
    v16bf pa = ldA(Pr + kl, Pr + kl + 16);

    // ---- out += P . v (K = 32 keys) ----
#pragma unroll
    for (int t = 0; t < 4; ++t) {
      v16bf vb = ldB(&Vlds[cur][t * 16 + l16][kb16]);
      acc[t] = WMMA_BF16(pa, vb, acc[t]);
    }

    wait_async0();     // this wave's next-tile async loads are done
    __syncthreads();   // all waves' loads done + all waves done reading buf[cur]
  }

  // ---- normalize and store [B,T,H] fp32 ----
#pragma unroll
  for (int r = 0; r < 8; ++r) {
    float inv = 1.0f / l[r];
    long long o = (long long)(b * Tc + t0 + r + half * 8) * Hc + l16;
    out[o +  0] = acc[0][r] * inv;
    out[o + 16] = acc[1][r] * inv;
    out[o + 32] = acc[2][r] * inv;
    out[o + 48] = acc[3][r] * inv;
  }
}

// ---------------------------------------------------------------------------
extern "C" void kernel_launch(void* const* d_in, const int* in_sizes, int n_in,
                              void* d_out, int out_size, void* d_ws, size_t ws_size,
                              hipStream_t stream) {
  const float* x    = (const float*)d_in[0];
  const int*   mask = (const int*)d_in[1];
  const float* Wk   = (const float*)d_in[2];
  const float* Wq   = (const float*)d_in[3];
  const float* Wv   = (const float*)d_in[4];
  float* out = (float*)d_out;

  char* ws = (char*)d_ws;
  __bf16* WT = (__bf16*)ws;                                   // 3*H*C bf16
  __bf16* qb = (__bf16*)(ws + (size_t)3 * Hc * Cc * 2);       // B*T*H bf16
  __bf16* kb = qb + (size_t)Bc * Tc * Hc;                     // B*T*H bf16
  __bf16* vT = kb + (size_t)Bc * Tc * Hc;                     // B*H*T bf16

  wt_kernel<<<(3 * Hc * Cc + 255) / 256, 256, 0, stream>>>(Wq, Wk, Wv, WT);
  proj_kernel<<<Bc * (Tc / 16), 128, 0, stream>>>(x, WT, qb, kb, vT);
  attn_kernel<<<Bc * (Tc / 16) / 4, 128, 0, stream>>>(qb, kb, vT, mask, out);
}
